// GATNet_64132451664026
// MI455X (gfx1250) — matrix operand order repaired and checked by reading, not compiled, not run on hardware
//
#include <hip/hip_runtime.h>

// ---------------------------------------------------------------------------
// GATNet on MI455X (gfx1250, wave32). Dense node GEMMs use
// v_wmma_f32_16x16x32_bf16 with async-to-LDS staged A tiles; edge passes are
// streaming kernels with L2-resident float atomics. Algebraic simplification:
// he = edge_attr@We is only ever contracted with a_e, so
// al_e = edge_attr @ (We . a_e) (a 16x2 weight), computed inline.
// ---------------------------------------------------------------------------

typedef __attribute__((ext_vector_type(16))) __bf16 v16bf;
typedef __attribute__((ext_vector_type(8)))  float  v8f;
typedef __attribute__((ext_vector_type(4)))  int    v4i;

#define GN 50000
#define GE 1600000
#define GH 2
#define GC 32
#define GHC 64
#define GIN 128
#define GED 16

#if defined(__has_builtin)
#if __has_builtin(__builtin_amdgcn_global_load_async_to_lds_b128) && \
    __has_builtin(__builtin_amdgcn_s_wait_asynccnt)
#define HAVE_ASYNC_LDS 1
#endif
#endif
#ifndef HAVE_ASYNC_LDS
#define HAVE_ASYNC_LDS 0
#endif

#if HAVE_ASYNC_LDS
typedef __attribute__((address_space(1))) v4i* gptr_v4i;  // global int4*
typedef __attribute__((address_space(3))) v4i* lptr_v4i;  // LDS int4*
#endif

__device__ __forceinline__ __bf16 f2bf(float f) {
    // native f32 -> bf16 conversion on gfx1250
    return (__bf16)f;
}

__device__ __forceinline__ void atomicMaxF32(float* addr, float val) {
    if (val >= 0.0f) atomicMax((int*)addr, __float_as_int(val));
    else             atomicMin((unsigned int*)addr, __float_as_uint(val));
}

// ---------------------------------------------------------------------------
// Dense node GEMM: Hout[M,64] = A[M,K] @ W[K,64], bf16 WMMA, f32 accumulate.
// Block = 128 threads = 4 waves; wave w computes 16x16 tile at cols [16w,16w+16).
// Block covers 16 rows; grid = M/16 (M = 50000 = 3125*16 exact).
// ---------------------------------------------------------------------------
__global__ __launch_bounds__(128)
void k_gemm_wmma(const float* __restrict__ A, const float* __restrict__ W,
                 float* __restrict__ Hout, int K) {
    __shared__ float sA[16 * 32];
    const int tid  = threadIdx.x;
    const int wave = tid >> 5;
    const int lane = tid & 31;
    const int row0 = blockIdx.x * 16;

    const int m   = lane & 15;                 // A row within tile
    const int kbA = (lane < 16) ? 0 : 16;      // A K-base (contiguous, matches B)
    const int n   = (wave << 4) + (lane & 15); // output column
    const int kbB = (lane < 16) ? 0 : 16;      // B K-base (contiguous, matches A)

    v8f acc = {0.f, 0.f, 0.f, 0.f, 0.f, 0.f, 0.f, 0.f};

    for (int kc = 0; kc < K; kc += 32) {
        // stage A tile [16 x 32] fp32 into LDS (16B per thread)
        {
            const int r  = tid >> 3;        // 0..15
            const int c4 = (tid & 7) << 2;  // 0,4,...,28
            const float* gsrc = &A[(size_t)(row0 + r) * K + kc + c4];
            float* ldst = &sA[r * 32 + c4];
#if HAVE_ASYNC_LDS
            // ASYNCcnt path: global_load_async_to_lds_b128
            __builtin_amdgcn_global_load_async_to_lds_b128(
                (gptr_v4i)(unsigned long long)gsrc,
                (lptr_v4i)(unsigned)(unsigned long long)ldst,
                0, 0);
            __builtin_amdgcn_s_wait_asynccnt(0);
#else
            const float4 v = *reinterpret_cast<const float4*>(gsrc);
            ldst[0] = v.x; ldst[1] = v.y; ldst[2] = v.z; ldst[3] = v.w;
#endif
        }
        __syncthreads();

        // A fragment: lanes 0-15 hold K=0..15, lanes 16-31 hold K=16..31 (matches B)
        v16bf a, b;
#pragma unroll
        for (int i = 0; i < 16; ++i) {
            const int k = kbA + i;
            a[i] = f2bf(sA[m * 32 + k]);
        }
        // B fragment: 32x16, lanes 0-15 hold K=0..15, lanes 16-31 hold K=16..31
#pragma unroll
        for (int i = 0; i < 16; ++i) {
            b[i] = f2bf(W[(size_t)(kc + kbB + i) * GHC + n]);
        }

        acc = __builtin_amdgcn_wmma_f32_16x16x32_bf16(
            false, a, false, b, (short)0, acc, false, false);
        __syncthreads();
    }

    // C/D layout: lanes 0-15 -> M=r, lanes 16-31 -> M=8+r
    const int mb = (lane < 16) ? 0 : 8;
#pragma unroll
    for (int r = 0; r < 8; ++r) {
        Hout[(size_t)(row0 + mb + r) * GHC + n] = acc[r];
    }
}

// ---------------------------------------------------------------------------
// Per-node attention logit halves: als/ald[n,h] = sum_c h[n,h,c]*a{s,d}[h,c]
// ---------------------------------------------------------------------------
__global__ void k_node_attn(const float* __restrict__ Hh,
                            const float* __restrict__ a_s,
                            const float* __restrict__ a_d,
                            float* __restrict__ als, float* __restrict__ ald) {
    const int idx = blockIdx.x * blockDim.x + threadIdx.x;
    if (idx >= GN * GH) return;
    const int n = idx >> 1, h = idx & 1;
    const float* row = Hh + (size_t)n * GHC + h * GC;
    const float* w1 = a_s + h * GC;
    const float* w2 = a_d + h * GC;
    float s = 0.f, d = 0.f;
#pragma unroll
    for (int c = 0; c < GC; ++c) { const float v = row[c]; s += v * w1[c]; d += v * w2[c]; }
    als[idx] = s;
    ald[idx] = d;
}

__global__ void k_init_nm(float* __restrict__ mbuf, float* __restrict__ denom) {
    const int idx = blockIdx.x * blockDim.x + threadIdx.x;
    if (idx >= GN * GH) return;
    mbuf[idx]  = -__builtin_inff();
    denom[idx] = 0.f;
}

// ---------------------------------------------------------------------------
// Fused: al_e inline (edge_attr . weff, weff[d,h]=sum_c We[d,h*C+c]*ae[h,c]),
// leaky_relu logits, segment max over dst (atomic max into L2-resident [N,2]).
// One thread per edge, both heads.
// ---------------------------------------------------------------------------
__global__ void k_edge_logits_max(const float* __restrict__ ea,
                                  const float* __restrict__ We,
                                  const float* __restrict__ ae,
                                  const int* __restrict__ src, const int* __restrict__ dst,
                                  const float* __restrict__ als, const float* __restrict__ ald,
                                  float* __restrict__ logits, float* __restrict__ mbuf) {
    __shared__ float weff[GED][GH];
    const int tid = threadIdx.x;
    if (tid < GED * GH) {
        const int d = tid >> 1, h = tid & 1;
        float s = 0.f;
#pragma unroll
        for (int c = 0; c < GC; ++c) s += We[d * GHC + h * GC + c] * ae[h * GC + c];
        weff[d][h] = s;
    }
    __syncthreads();
    const int e = blockIdx.x * blockDim.x + tid;
    if (e >= GE) return;
    const float* row = ea + (size_t)e * GED;
    float s0 = 0.f, s1 = 0.f;
#pragma unroll
    for (int d = 0; d < GED; ++d) {
        const float v = row[d];
        s0 += v * weff[d][0];
        s1 += v * weff[d][1];
    }
    const int s = src[e], d = dst[e];
    float v0 = als[s * 2 + 0] + ald[d * 2 + 0] + s0;
    float v1 = als[s * 2 + 1] + ald[d * 2 + 1] + s1;
    v0 = (v0 > 0.f) ? v0 : 0.2f * v0;   // leaky_relu(0.2)
    v1 = (v1 > 0.f) ? v1 : 0.2f * v1;
    logits[(size_t)e * 2 + 0] = v0;
    logits[(size_t)e * 2 + 1] = v1;
    atomicMaxF32(&mbuf[d * 2 + 0], v0);
    atomicMaxF32(&mbuf[d * 2 + 1], v1);
}

// e = exp(logit - m[dst]); segment sum into denom (in-place logits -> e)
__global__ void k_edge_exp_sum(const int* __restrict__ dst,
                               float* __restrict__ evals,
                               const float* __restrict__ mbuf,
                               float* __restrict__ denom) {
    const int idx = blockIdx.x * blockDim.x + threadIdx.x;
    if (idx >= GE * GH) return;
    const int e = idx >> 1, h = idx & 1;
    const int d = dst[e];
    const float ev = __expf(evals[idx] - mbuf[d * 2 + h]);
    evals[idx] = ev;
    atomicAdd(&denom[d * 2 + h], ev);
}

__global__ void k_init_out(float* __restrict__ outb, const float* __restrict__ bias) {
    const int idx = blockIdx.x * blockDim.x + threadIdx.x;
    if (idx >= GN * GHC) return;
    outb[idx] = bias[idx & (GHC - 1)];
}

// out[dst, c] += h[src, c] * alpha[e, c/32]; 12.8MB accumulator lives in L2
__global__ void k_edge_aggregate(const int* __restrict__ src, const int* __restrict__ dst,
                                 const float* __restrict__ Hh,
                                 const float* __restrict__ evals,
                                 const float* __restrict__ denom,
                                 float* __restrict__ outb) {
    const long long idx = (long long)blockIdx.x * blockDim.x + threadIdx.x;
    if (idx >= (long long)GE * GHC) return;
    const int e = (int)(idx >> 6);
    const int c = (int)(idx & 63);
    const int h = c >> 5;
    if (c == 0 && e + 4096 < GE) {      // stream-ahead hint -> global_prefetch_b8
        __builtin_prefetch(&src[e + 4096], 0, 0);
        __builtin_prefetch(&dst[e + 4096], 0, 0);
    }
    const int s = src[e], d = dst[e];
    const float alpha = evals[e * 2 + h] / (denom[d * 2 + h] + 1e-16f);
    atomicAdd(&outb[(size_t)d * GHC + c], Hh[(size_t)s * GHC + c] * alpha);
}

__global__ void k_node_elu(float* __restrict__ buf) {
    const int idx = blockIdx.x * blockDim.x + threadIdx.x;
    if (idx >= GN * GHC) return;
    const float v = buf[idx];
    buf[idx] = (v > 0.f) ? v : (__expf(v) - 1.0f);
}

// logit = h3 @ Wlin + blin; out[n] = sigmoid(logit); concat row tail = logit
__global__ void k_final_head(const float* __restrict__ h3,
                             const float* __restrict__ Wlin,
                             const float* __restrict__ blin,
                             float* __restrict__ out) {
    const int n = blockIdx.x * blockDim.x + threadIdx.x;
    if (n >= GN) return;
    const float* row = h3 + (size_t)n * GHC;
    float s = 0.f;
#pragma unroll
    for (int c = 0; c < GHC; ++c) s += row[c] * Wlin[c];
    const float logit = s + blin[0];
    out[n] = 1.0f / (1.0f + __expf(-logit));
    out[(size_t)65 * GN + (size_t)n * 65 + 64] = logit;
}

// feat copies: out[N + n*64 + c] and out[65N + n*65 + c]
__global__ void k_final_feat(const float* __restrict__ feat, float* __restrict__ out) {
    const int idx = blockIdx.x * blockDim.x + threadIdx.x;
    if (idx >= GN * GHC) return;
    const int n = idx >> 6, c = idx & 63;
    const float f = feat[idx];
    out[(size_t)GN + idx] = f;
    out[(size_t)65 * GN + (size_t)n * 65 + c] = f;
}

// ---------------------------------------------------------------------------
extern "C" void kernel_launch(void* const* d_in, const int* in_sizes, int n_in,
                              void* d_out, int out_size, void* d_ws, size_t ws_size,
                              hipStream_t stream) {
    (void)in_sizes; (void)n_in; (void)out_size; (void)ws_size;
    const float* x         = (const float*)d_in[0];
    const int*   ei        = (const int*)d_in[1];
    const float* edge_attr = (const float*)d_in[2];
    const float* W[3]  = {(const float*)d_in[3],  (const float*)d_in[9],  (const float*)d_in[15]};
    const float* We[3] = {(const float*)d_in[4],  (const float*)d_in[10], (const float*)d_in[16]};
    const float* As[3] = {(const float*)d_in[5],  (const float*)d_in[11], (const float*)d_in[17]};
    const float* Ad[3] = {(const float*)d_in[6],  (const float*)d_in[12], (const float*)d_in[18]};
    const float* Ae[3] = {(const float*)d_in[7],  (const float*)d_in[13], (const float*)d_in[19]};
    const float* Bi[3] = {(const float*)d_in[8],  (const float*)d_in[14], (const float*)d_in[20]};
    const float* Wlin  = (const float*)d_in[21];
    const float* blin  = (const float*)d_in[22];
    const int* src = ei;
    const int* dst = ei + GE;
    float* out = (float*)d_out;

    // workspace layout (floats); total ~13.2M floats (~53 MB)
    float* B0    = (float*)d_ws;            // [N,64] transformed features (per layer)
    float* B1    = B0   + (size_t)GN * GHC; // layer-1 / layer-3 aggregated output
    float* B2    = B1   + (size_t)GN * GHC; // layer-2 output = feat (kept)
    float* als   = B2   + (size_t)GN * GHC; // [N,2]
    float* ald   = als  + (size_t)GN * GH;
    float* mbuf  = ald  + (size_t)GN * GH;
    float* denom = mbuf + (size_t)GN * GH;
    float* evals = denom + (size_t)GN * GH; // [E,2] logits then exp-values

    const int TB = 256;
    const dim3 gNodeH((GN * GH + TB - 1) / TB);
    const dim3 gNodeC((GN * GHC + TB - 1) / TB);
    const dim3 gEdge((GE + TB - 1) / TB);
    const dim3 gEdgeH((GE * GH + TB - 1) / TB);
    const dim3 gEdgeC((unsigned)(((long long)GE * GHC + TB - 1) / TB));

    const float* in = x;
    int K = GIN;
    float* outs[3] = {B1, B2, B1};
    for (int l = 0; l < 3; ++l) {
        k_gemm_wmma<<<GN / 16, 128, 0, stream>>>(in, W[l], B0, K);
        k_node_attn<<<gNodeH, TB, 0, stream>>>(B0, As[l], Ad[l], als, ald);
        k_init_nm<<<gNodeH, TB, 0, stream>>>(mbuf, denom);
        k_edge_logits_max<<<gEdge, TB, 0, stream>>>(edge_attr, We[l], Ae[l],
                                                    src, dst, als, ald, evals, mbuf);
        k_edge_exp_sum<<<gEdgeH, TB, 0, stream>>>(dst, evals, mbuf, denom);
        k_init_out<<<gNodeC, TB, 0, stream>>>(outs[l], Bi[l]);
        k_edge_aggregate<<<gEdgeC, TB, 0, stream>>>(src, dst, B0, evals, denom, outs[l]);
        k_node_elu<<<gNodeC, TB, 0, stream>>>(outs[l]);
        in = outs[l];
        K = GHC;
    }

    k_final_head<<<(GN + TB - 1) / TB, TB, 0, stream>>>(B1, Wlin, blin, out);
    k_final_feat<<<gNodeC, TB, 0, stream>>>(B2, out);
}
